// GAT_50002009260138
// MI455X (gfx1250) — compile-verified, matrix operand order checked by reading, not run on hardware
//
#include <hip/hip_runtime.h>
#include <hip/hip_bf16.h>

typedef unsigned short u16;
typedef __attribute__((ext_vector_type(8)))  float        v8f;
typedef __attribute__((ext_vector_type(16))) __bf16       v16bf;
typedef __attribute__((ext_vector_type(4)))  unsigned int u32x4;
typedef __attribute__((ext_vector_type(8)))  int          i32x8;
typedef __attribute__((ext_vector_type(4)))  int          i32x4;

#if defined(__has_builtin)
#  if __has_builtin(__builtin_amdgcn_tensor_load_to_lds)
#    define HAVE_TDM 1
#  else
#    define HAVE_TDM 0
#  endif
#else
#  define HAVE_TDM 0
#endif

union Frag { u32x4 u[2]; v16bf v; };

__device__ __forceinline__ u16 f2bf(float f) {
  unsigned u = __float_as_uint(f);
  u += 0x7FFFu + ((u >> 16) & 1u);      // round-to-nearest-even
  return (u16)(u >> 16);
}
__device__ __forceinline__ unsigned f2ord(float f) {  // monotone float->uint
  unsigned u = __float_as_uint(f);
  return (u & 0x80000000u) ? ~u : (u | 0x80000000u);
}
__device__ __forceinline__ float ord2f(unsigned u) {
  return (u & 0x80000000u) ? __uint_as_float(u & 0x7FFFFFFFu) : __uint_as_float(~u);
}
// low 32 bits of a generic LDS pointer == byte offset in the wave's LDS window
__device__ __forceinline__ unsigned lds_off(const void* p) {
  return (unsigned)(size_t)p;
}

// ---------------- elementwise helpers ----------------
__global__ void k_convert_bf16(const float* __restrict__ src, u16* __restrict__ dst, size_t n) {
  size_t i = (size_t)blockIdx.x * blockDim.x + threadIdx.x;
  if (i < n) dst[i] = f2bf(src[i]);
}
__global__ void k_zero_f32(float* __restrict__ p, size_t n) {
  size_t i = (size_t)blockIdx.x * blockDim.x + threadIdx.x;
  if (i < n) p[i] = 0.f;
}
__global__ void k_init_attn(unsigned* __restrict__ mord, float* __restrict__ denom, int n) {
  int i = blockIdx.x * blockDim.x + threadIdx.x;
  if (i < n) { mord[i] = 0u; denom[i] = 0.f; }  // ord 0 acts as -inf sentinel
}
__global__ void k_bias_relu(float* __restrict__ x, const float* __restrict__ b,
                            size_t rows, int cols) {
  size_t i = (size_t)blockIdx.x * blockDim.x + threadIdx.x;
  if (i >= rows * (size_t)cols) return;
  int c = (int)(i % cols);
  float v = x[i] + b[c];
  x[i] = v > 0.f ? v : 0.f;
}

// Transpose + convert: src [K][N] f32 row-major  ->  dst [N][K] bf16
__global__ void k_transpose_bf16(const float* __restrict__ src, u16* __restrict__ dst,
                                 int K, int N) {
  __shared__ float tile[32][33];
  int k0 = blockIdx.y * 32, n0 = blockIdx.x * 32;
  int tx = threadIdx.x, ty = threadIdx.y;        // block (32,8)
  #pragma unroll
  for (int i = ty; i < 32; i += 8) {
    int k = k0 + i, n = n0 + tx;
    tile[i][tx] = (k < K && n < N) ? src[(size_t)k * N + n] : 0.f;
  }
  __syncthreads();
  #pragma unroll
  for (int i = ty; i < 32; i += 8) {
    int n = n0 + i, k = k0 + tx;
    if (n < N && k < K) dst[(size_t)n * K + k] = f2bf(tile[tx][i]);
  }
}

// ---------------- WMMA bf16 GEMM: C[M,N] = A[M,K] * Bt[N,K]^T (+bias, relu) ----------------
// Double-buffered pipeline: next K-slab is DMA'd (async-to-LDS for A, TDM descriptor for B)
// while WMMAs consume the current slab. One barrier per K-step.
#define BM 128
#define BN 128
#define BK 32
#define LDSK 40   // padded row stride (u16) -> 80B; TDM pad 4 DW per 16 DW reproduces this

__global__ __launch_bounds__(256)
void k_gemm_bf16(const u16* __restrict__ A, const u16* __restrict__ Bt,
                 float* __restrict__ C, const float* __restrict__ bias,
                 int M, int N, int K, int relu) {
  __shared__ __align__(16) u16 lA[2][BM * LDSK];
  __shared__ __align__(16) u16 lB[2][BN * LDSK];

  const int tid   = threadIdx.x;
  const int lane  = tid & 31;
  const int wave  = tid >> 5;       // 0..7
  const int waveM = wave >> 1;      // 0..3 : 32 rows each
  const int waveN = wave & 1;       // 0..1 : 64 cols each
  const int tileM0 = blockIdx.y * BM;
  const int tileN0 = blockIdx.x * BN;

  const int r   = tid >> 1;         // 0..127 : tile row to load
  const int seg = tid & 1;          // 16-element half of the 32-wide K slab
  const int laneM = lane & 15;
  const int laneK = (lane >> 4) * 8; // K sub-base per 16-bit A/B fragment layout

  // --- stage A slab: per-lane async global->LDS DMA (ASYNCcnt). OOB rows are simply
  // skipped by EXEC; garbage only reaches store-guarded C rows (K is full). ---
  auto stageA = [&](int buf, int kt) {
    int gRow = tileM0 + r;
    unsigned la = lds_off(&lA[buf][r * LDSK + seg * 16]);
    if (gRow < M) {
      const u16* g = A + (size_t)gRow * K + kt + seg * 16;
      asm volatile("global_load_async_to_lds_b128 %0, %1, off"
                   :: "v"(la), "v"(g) : "memory");
    }
  };

  // --- stage B slab: one Tensor Data Mover descriptor (TENSORcnt), issued by wave 0.
  // 2D tile: 32 (K) x 128 (N rows), data_size=2B, LDS padding 4 DW per 16 DW -> pitch 40 u16.
  // tensor_dim1 = N - tileN0 so the TDM zero-fills out-of-range N rows itself. ---
  auto stageB = [&](int buf, int kt) {
#if HAVE_TDM
    if (wave == 0) {
      unsigned long long ga = (unsigned long long)(size_t)(Bt + (size_t)tileN0 * K + kt);
      unsigned ldsb  = lds_off(&lB[buf][0]);
      unsigned rows  = (unsigned)(N - tileN0);
      unsigned k_rem = (unsigned)(K - kt);
      u32x4 g0;
      g0[0] = 1u;                                                   // count=1 (valid D#)
      g0[1] = ldsb;                                                 // lds_addr
      g0[2] = (unsigned)ga;                                         // global_addr[31:0]
      g0[3] = (unsigned)((ga >> 32) & 0x01FFFFFFull) | (2u << 30);  // addr[56:32], type=2
      i32x8 g1;
      g1[0] = (int)((1u << 16) | (1u << 20) | (3u << 22) | (3u << 25)); // 2B, pad_en, 16DW/4DW
      g1[1] = (int)((k_rem & 0xFFFFu) << 16);                       // tensor_dim0[15:0]
      g1[2] = (int)(((k_rem >> 16) & 0xFFFFu) | ((rows & 0xFFFFu) << 16)); // dim0 hi | dim1 lo
      g1[3] = (int)(((rows >> 16) & 0xFFFFu) | (32u << 16));        // dim1 hi | tile_dim0=32
      g1[4] = (int)128u;                                            // tile_dim1=128, tile_dim2=0
      g1[5] = (int)(unsigned)K;                                     // tensor_dim0_stride[31:0]
      g1[6] = 0;                                                    // stride hi, dim1_stride=0
      g1[7] = 0;
      i32x4 zz4 = {0, 0, 0, 0};
      i32x8 zz8 = {0, 0, 0, 0, 0, 0, 0, 0};
      // 6-arg toolchain variant: (g0, g1, g2, g3, extra, cpol)
      __builtin_amdgcn_tensor_load_to_lds(g0, g1, zz4, zz4, zz8, 0);
    }
#else
    int gN = tileN0 + r;
    unsigned lb = lds_off(&lB[buf][r * LDSK + seg * 16]);
    if (gN < N) {
      const u16* g = Bt + (size_t)gN * K + kt + seg * 16;
      asm volatile("global_load_async_to_lds_b128 %0, %1, off"
                   :: "v"(lb), "v"(g) : "memory");
    }
#endif
  };

  auto wait_dma = [&]() {
    asm volatile("s_wait_asynccnt 0x0" ::: "memory");
#if HAVE_TDM
    __builtin_amdgcn_s_wait_tensorcnt(0);
#endif
  };

  const v8f vzero = {0.f, 0.f, 0.f, 0.f, 0.f, 0.f, 0.f, 0.f};
  v8f acc[2][4];
  #pragma unroll
  for (int mi = 0; mi < 2; ++mi)
    #pragma unroll
    for (int nj = 0; nj < 4; ++nj) acc[mi][nj] = vzero;

  // prologue: fill buffer 0
  stageA(0, 0);
  stageB(0, 0);
  wait_dma();
  __syncthreads();

  int buf = 0;
  for (int kt = 0; kt < K; kt += BK) {
    int nk = kt + BK;
    if (nk < K) {               // kick DMA for the next slab into the other buffer
      stageA(buf ^ 1, nk);
      stageB(buf ^ 1, nk);
    }

    // ---- fragment loads (two b128 per fragment; 16-bit A/B lane layout) ----
    Frag aF[2], bF[4];
    #pragma unroll
    for (int mi = 0; mi < 2; ++mi) {
      int row = waveM * 32 + mi * 16 + laneM;
      const u32x4* p = reinterpret_cast<const u32x4*>(&lA[buf][row * LDSK + laneK]);
      aF[mi].u[0] = p[0];          // K = laneK .. laneK+7
      aF[mi].u[1] = p[2];          // K = laneK+16 .. laneK+23
    }
    #pragma unroll
    for (int nj = 0; nj < 4; ++nj) {
      int col = waveN * 64 + nj * 16 + laneM;
      const u32x4* p = reinterpret_cast<const u32x4*>(&lB[buf][col * LDSK + laneK]);
      bF[nj].u[0] = p[0];
      bF[nj].u[1] = p[2];
    }
    // ---- 8 WMMAs per wave per K-step ----
    #pragma unroll
    for (int mi = 0; mi < 2; ++mi)
      #pragma unroll
      for (int nj = 0; nj < 4; ++nj)
        acc[mi][nj] = __builtin_amdgcn_wmma_f32_16x16x32_bf16(
            false, aF[mi].v, false, bF[nj].v, (short)0, acc[mi][nj], false, false);

    if (nk < K) {
      wait_dma();                // this wave's DMA into buf^1 has landed
      __syncthreads();           // everyone done reading buf and done writing buf^1
    }
    buf ^= 1;
  }

  // ---- epilogue: lanes 0-15: N=lane, M=r; lanes 16-31: N=lane-16, M=8+r ----
  const int mHalf = (lane >> 4) * 8;
  const int nIdx  = lane & 15;
  #pragma unroll
  for (int mi = 0; mi < 2; ++mi)
    #pragma unroll
    for (int nj = 0; nj < 4; ++nj) {
      int col = tileN0 + waveN * 64 + nj * 16 + nIdx;
      if (col >= N) continue;
      float bv = bias ? bias[col] : 0.f;
      #pragma unroll
      for (int rr = 0; rr < 8; ++rr) {
        int row = tileM0 + waveM * 32 + mi * 16 + mHalf + rr;
        if (row < M) {
          float v = acc[mi][nj][rr] + bv;
          if (relu) v = v > 0.f ? v : 0.f;
          C[(size_t)row * N + col] = v;
        }
      }
    }
}

// ---------------- attention pieces ----------------
// one wave per (node, head): both alpha_src and alpha_dst dot products
__global__ void k_attn_dots(const float* __restrict__ xl,
                            const float* __restrict__ a_s, const float* __restrict__ a_d,
                            float* __restrict__ out_s, float* __restrict__ out_d,
                            int Nn, int Hh, int Cc) {
  int g = (blockIdx.x * blockDim.x + threadIdx.x) >> 5;   // wave id (uniform in wave)
  int lane = threadIdx.x & 31;
  if (g >= Nn * Hh) return;
  int n = g / Hh, h = g % Hh;
  const float* xp = xl + (size_t)n * Hh * Cc + (size_t)h * Cc;
  const float* as = a_s + (size_t)h * Cc;
  const float* ad = a_d + (size_t)h * Cc;
  float s1 = 0.f, s2 = 0.f;
  for (int c = lane; c < Cc; c += 32) {
    float v = xp[c];
    s1 += v * as[c];
    s2 += v * ad[c];
  }
  #pragma unroll
  for (int off = 16; off; off >>= 1) {
    s1 += __shfl_down(s1, off);
    s2 += __shfl_down(s2, off);
  }
  if (lane == 0) { out_s[g] = s1; out_d[g] = s2; }
}

// per (edge, head): leaky-relu score + segment max (ordered-uint atomicMax)
__global__ void k_edge_scores(const int* __restrict__ ei, int E, int ET, int Hh,
                              const float* __restrict__ as, const float* __restrict__ ad,
                              float* __restrict__ eval, unsigned* __restrict__ mord) {
  int idx = blockIdx.x * blockDim.x + threadIdx.x;
  if (idx >= ET * Hh) return;
  int e = idx / Hh, h = idx % Hh;
  int s, d;
  if (e < E) { s = ei[e]; d = ei[E + e]; } else { s = d = e - E; }  // self loops appended
  float v = as[s * Hh + h] + ad[d * Hh + h];
  v = v > 0.f ? v : 0.2f * v;
  eval[idx] = v;
  atomicMax(&mord[d * Hh + h], f2ord(v));
}

// per (edge, head): exp(e - max) + segment sum
__global__ void k_edge_exp(const int* __restrict__ ei, int E, int ET, int Hh,
                           float* __restrict__ eval, const unsigned* __restrict__ mord,
                           float* __restrict__ denom) {
  int idx = blockIdx.x * blockDim.x + threadIdx.x;
  if (idx >= ET * Hh) return;
  int e = idx / Hh, h = idx % Hh;
  int d = (e < E) ? ei[E + e] : e - E;
  float x = __expf(eval[idx] - ord2f(mord[d * Hh + h]));
  eval[idx] = x;
  atomicAdd(&denom[d * Hh + h], x);
}

// one block per edge: agg[dst] += (ex/denom[dst]) * xl[src]   (bandwidth-bound phase)
__global__ void k_aggregate(const int* __restrict__ ei, int E, int Hh, int Cc,
                            const float* __restrict__ eval, const float* __restrict__ denom,
                            const float* __restrict__ xl, float* __restrict__ agg) {
  int e = blockIdx.x;
  int s, d;
  if (e < E) { s = ei[e]; d = ei[E + e]; } else { s = d = e - E; }
  int HC = Hh * Cc;
  for (int idx = threadIdx.x; idx < HC; idx += blockDim.x) {
    int h = idx / Cc;
    float alpha = eval[e * Hh + h] / denom[d * Hh + h];
    atomicAdd(&agg[(size_t)d * HC + idx], alpha * xl[(size_t)s * HC + idx]);
  }
}

// ---------------- launcher ----------------
static inline unsigned cdiv(size_t a, size_t b) { return (unsigned)((a + b - 1) / b); }

extern "C" void kernel_launch(void* const* d_in, const int* in_sizes, int n_in,
                              void* d_out, int out_size, void* d_ws, size_t ws_size,
                              hipStream_t stream) {
  (void)n_in; (void)out_size; (void)ws_size;
  const int F_IN = 2048, H = 4, C1 = 1024, HC1 = H * C1, C2 = 64;
  const int NPG = 420, OUT = 18;
  const int Nn = in_sizes[0] / F_IN;   // 3360
  const int E  = in_sizes[1] / 2;      // 53760
  const int ET = E + Nn;               // edges + self loops
  const int Bb = Nn / NPG;             // 8 graphs
  const int KFC1 = NPG * C2;           // 26880

  const float* x   = (const float*)d_in[0];
  const int*   ei  = (const int*)d_in[1];
  const float* W1  = (const float*)d_in[2];
  const float* as1 = (const float*)d_in[3];
  const float* ad1 = (const float*)d_in[4];
  const float* b1  = (const float*)d_in[5];
  const float* W2  = (const float*)d_in[6];
  const float* as2 = (const float*)d_in[7];
  const float* ad2 = (const float*)d_in[8];
  const float* b2  = (const float*)d_in[9];
  const float* wf1 = (const float*)d_in[10];
  const float* bfc1= (const float*)d_in[11];
  const float* wf2 = (const float*)d_in[12];
  const float* bfc2= (const float*)d_in[13];
  const float* wf3 = (const float*)d_in[14];
  const float* bfc3= (const float*)d_in[15];
  const float* wf4 = (const float*)d_in[16];
  const float* bfc4= (const float*)d_in[17];
  float* out = (float*)d_out;

  char* ws = (char*)d_ws;
  size_t off = 0;
  auto take = [&](size_t bytes) -> void* {
    void* p = ws + off;
    off = (off + bytes + 255) & ~(size_t)255;
    return p;
  };

  // phase-1 buffers (peak ~142 MB with reuse below)
  u16*   xbf   = (u16*)  take((size_t)Nn * F_IN * 2);   // region A
  u16*   W1t   = (u16*)  take((size_t)HC1 * F_IN * 2);  // region B
  size_t offC  = off;
  float* xl1   = (float*)take((size_t)Nn * HC1 * 4);    // region C
  float* agg1  = (float*)take((size_t)Nn * HC1 * 4);    // region D
  float* asrc1 = (float*)take((size_t)Nn * H * 4);
  float* adst1 = (float*)take((size_t)Nn * H * 4);
  unsigned* mord1 = (unsigned*)take((size_t)Nn * H * 4);
  float* den1  = (float*)take((size_t)Nn * H * 4);
  float* ev1   = (float*)take((size_t)ET * H * 4);

  // region A+B reused after GEMM1 consumers are done (27.5 MB <= 30.6 MB)
  u16* h1bf = (u16*)ws;

  // region C reused after layer-1 aggregation (xl1 dead); ~17 MB << 55 MB
  size_t off2 = offC;
  auto take2 = [&](size_t bytes) -> void* {
    void* p = ws + off2;
    off2 = (off2 + bytes + 255) & ~(size_t)255;
    return p;
  };
  u16*   W2t   = (u16*)  take2((size_t)C2 * HC1 * 2);
  float* xl2   = (float*)take2((size_t)Nn * C2 * 4);
  float* asrc2 = (float*)take2((size_t)Nn * 4);
  float* adst2 = (float*)take2((size_t)Nn * 4);
  unsigned* mord2 = (unsigned*)take2((size_t)Nn * 4);
  float* den2  = (float*)take2((size_t)Nn * 4);
  float* ev2   = (float*)take2((size_t)ET * 4);
  float* agg2  = (float*)take2((size_t)Nn * C2 * 4);
  u16*   h2bf  = (u16*)  take2((size_t)Nn * C2 * 2);
  u16*   f1wt  = (u16*)  take2((size_t)256 * KFC1 * 2);
  u16*   f2wt  = (u16*)  take2((size_t)128 * 256 * 2);
  u16*   f3wt  = (u16*)  take2((size_t)64 * 128 * 2);
  u16*   f4wt  = (u16*)  take2((size_t)OUT * 64 * 2);
  float* f1o   = (float*)take2((size_t)Bb * 256 * 4);
  u16*   f1b   = (u16*)  take2((size_t)Bb * 256 * 2);
  float* f2o   = (float*)take2((size_t)Bb * 128 * 4);
  u16*   f2b   = (u16*)  take2((size_t)Bb * 128 * 2);
  float* f3o   = (float*)take2((size_t)Bb * 64 * 4);
  u16*   f3b   = (u16*)  take2((size_t)Bb * 64 * 2);

  const dim3 tb(32, 8);
  #define GEMM(A_, Bt_, C_, bias_, M_, N_, K_, relu_) \
    k_gemm_bf16<<<dim3(cdiv((N_),BN), cdiv((M_),BM)), 256, 0, stream>>>( \
        A_, Bt_, C_, bias_, M_, N_, K_, relu_)

  // ===== GAT layer 1 =====
  k_convert_bf16<<<cdiv((size_t)Nn*F_IN,256),256,0,stream>>>(x, xbf, (size_t)Nn*F_IN);
  k_transpose_bf16<<<dim3(cdiv(HC1,32), cdiv(F_IN,32)), tb, 0, stream>>>(W1, W1t, F_IN, HC1);
  GEMM(xbf, W1t, xl1, (const float*)nullptr, Nn, HC1, F_IN, 0);     // xl1 = x @ W1

  k_attn_dots<<<cdiv((size_t)Nn*H*32,256),256,0,stream>>>(xl1, as1, ad1, asrc1, adst1, Nn, H, C1);
  k_init_attn<<<cdiv((size_t)Nn*H,256),256,0,stream>>>(mord1, den1, Nn*H);
  k_zero_f32<<<cdiv((size_t)Nn*HC1,256),256,0,stream>>>(agg1, (size_t)Nn*HC1);
  k_edge_scores<<<cdiv((size_t)ET*H,256),256,0,stream>>>(ei, E, ET, H, asrc1, adst1, ev1, mord1);
  k_edge_exp   <<<cdiv((size_t)ET*H,256),256,0,stream>>>(ei, E, ET, H, ev1, mord1, den1);
  k_aggregate  <<<ET, 256, 0, stream>>>(ei, E, H, C1, ev1, den1, xl1, agg1);
  k_bias_relu  <<<cdiv((size_t)Nn*HC1,256),256,0,stream>>>(agg1, b1, (size_t)Nn, HC1);  // h1

  // ===== GAT layer 2 =====
  k_convert_bf16<<<cdiv((size_t)Nn*HC1,256),256,0,stream>>>(agg1, h1bf, (size_t)Nn*HC1);
  k_transpose_bf16<<<dim3(cdiv(C2,32), cdiv(HC1,32)), tb, 0, stream>>>(W2, W2t, HC1, C2);
  GEMM(h1bf, W2t, xl2, (const float*)nullptr, Nn, C2, HC1, 0);      // xl2 = h1 @ W2

  k_attn_dots<<<cdiv((size_t)Nn*32,256),256,0,stream>>>(xl2, as2, ad2, asrc2, adst2, Nn, 1, C2);
  k_init_attn<<<cdiv((size_t)Nn,256),256,0,stream>>>(mord2, den2, Nn);
  k_zero_f32<<<cdiv((size_t)Nn*C2,256),256,0,stream>>>(agg2, (size_t)Nn*C2);
  k_edge_scores<<<cdiv((size_t)ET,256),256,0,stream>>>(ei, E, ET, 1, asrc2, adst2, ev2, mord2);
  k_edge_exp   <<<cdiv((size_t)ET,256),256,0,stream>>>(ei, E, ET, 1, ev2, mord2, den2);
  k_aggregate  <<<ET, 64, 0, stream>>>(ei, E, 1, C2, ev2, den2, xl2, agg2);
  k_bias_relu  <<<cdiv((size_t)Nn*C2,256),256,0,stream>>>(agg2, b2, (size_t)Nn, C2);    // h2

  // ===== FC head: [8, 26880] -> 256 -> 128 -> 64 -> 18 =====
  k_convert_bf16<<<cdiv((size_t)Nn*C2,256),256,0,stream>>>(agg2, h2bf, (size_t)Nn*C2);
  k_transpose_bf16<<<dim3(cdiv(256,32), cdiv(KFC1,32)), tb, 0, stream>>>(wf1, f1wt, KFC1, 256);
  GEMM(h2bf, f1wt, f1o, bfc1, Bb, 256, KFC1, 1);
  k_convert_bf16<<<cdiv((size_t)Bb*256,256),256,0,stream>>>(f1o, f1b, (size_t)Bb*256);
  k_transpose_bf16<<<dim3(cdiv(128,32), cdiv(256,32)), tb, 0, stream>>>(wf2, f2wt, 256, 128);
  GEMM(f1b, f2wt, f2o, bfc2, Bb, 128, 256, 1);
  k_convert_bf16<<<cdiv((size_t)Bb*128,256),256,0,stream>>>(f2o, f2b, (size_t)Bb*128);
  k_transpose_bf16<<<dim3(cdiv(64,32), cdiv(128,32)), tb, 0, stream>>>(wf3, f3wt, 128, 64);
  GEMM(f2b, f3wt, f3o, bfc3, Bb, 64, 128, 1);
  k_convert_bf16<<<cdiv((size_t)Bb*64,256),256,0,stream>>>(f3o, f3b, (size_t)Bb*64);
  k_transpose_bf16<<<dim3(cdiv(OUT,32), cdiv(64,32)), tb, 0, stream>>>(wf4, f4wt, 64, OUT);
  GEMM(f3b, f4wt, out, bfc4, Bb, OUT, 64, 0);
  #undef GEMM
}
